// LigerFusedLinearCPOLoss_70325794505338
// MI455X (gfx1250) — compile-verified, matrix operand order checked by reading, not run on hardware
//
#include <hip/hip_runtime.h>
#include <hip/hip_bf16.h>
#include <math.h>

// Problem constants (from reference)
#define B2 8
#define T_LEN 1024
#define H_DIM 2048
#define V_DIM 32000
#define M_TOK (B2 * T_LEN)        // 8192 tokens
#define IGNORE_INDEX (-100)
#define BETA_C 0.1f
#define ALPHA_C 1.0f

// LSE GEMM tiling
#define TILE_M 32                 // tokens per block
#define TILE_N 64                 // logit columns per n-iteration (4 n-waves x 16)
#define LDA_H 2056                // padded LDS row stride (bf16 elems), 16B-aligned rows

typedef __attribute__((ext_vector_type(16))) __bf16 bf16x16;
typedef __attribute__((ext_vector_type(8)))  float  f32x8;

union BF16Frag {
    bf16x16        v;
    unsigned short u[16];
    uint4          q[2];
};

__device__ __forceinline__ unsigned short f2bf_bits(float f) {
    // round-to-nearest-even f32 -> bf16 bit pattern
    unsigned int x = __float_as_uint(f);
    unsigned int r = x + 0x7fffu + ((x >> 16) & 1u);
    return (unsigned short)(r >> 16);
}

// ---------------------------------------------------------------------------
// Kernel 0: one-shot W f32 -> bf16 conversion into workspace (bandwidth-bound,
// ~400 MB of traffic ~= 17 us at 23.3 TB/s; makes the GEMM WMMA-bound and lets
// bf16 W (131 MB) live entirely in the 192 MB L2).
// ---------------------------------------------------------------------------
__global__ __launch_bounds__(256)
void convert_w_kernel(const float* __restrict__ W, unsigned short* __restrict__ Wbf) {
    const size_t n4 = (size_t)V_DIM * H_DIM / 4;   // float4 count
    const float4* src = (const float4*)W;
    uint2* dst = (uint2*)Wbf;
    size_t stride = (size_t)gridDim.x * blockDim.x;
    for (size_t i = (size_t)blockIdx.x * blockDim.x + threadIdx.x; i < n4; i += stride) {
        float4 f = src[i];
        uint2 p;
        p.x = (unsigned int)f2bf_bits(f.x) | ((unsigned int)f2bf_bits(f.y) << 16);
        p.y = (unsigned int)f2bf_bits(f.z) | ((unsigned int)f2bf_bits(f.w) << 16);
        dst[i] = p;
    }
}

// ---------------------------------------------------------------------------
// Kernel 1: per-token target logit  tgt[i] = dot(X[i,:], W[target[i],:]) + b[t]
// ---------------------------------------------------------------------------
__global__ __launch_bounds__(256)
void target_logit_kernel(const float* __restrict__ W,
                         const float* __restrict__ X,
                         const int*   __restrict__ target,
                         const float* __restrict__ bias,
                         float* __restrict__ tgt_out) {
    const int lane  = threadIdx.x & 31;
    const int wid   = threadIdx.x >> 5;
    const int token = blockIdx.x * 8 + wid;
    if (token >= M_TOK) return;

    int t = target[token];
    const bool valid = (t != IGNORE_INDEX);
    int tc = valid ? t : 0;
    if (tc < 0) tc = 0;
    if (tc >= V_DIM) tc = V_DIM - 1;

    const float4* xr = (const float4*)(X + (size_t)token * H_DIM);
    const float4* wr = (const float4*)(W + (size_t)tc    * H_DIM);

    float dot = 0.0f;
    #pragma unroll 4
    for (int j = 0; j < H_DIM / 128; ++j) {
        float4 a = xr[j * 32 + lane];
        float4 b = wr[j * 32 + lane];
        dot += a.x * b.x + a.y * b.y + a.z * b.z + a.w * b.w;
    }
    #pragma unroll
    for (int off = 16; off > 0; off >>= 1)
        dot += __shfl_xor(dot, off, 32);

    if (lane == 0)
        tgt_out[token] = valid ? (dot + bias[tc]) : 0.0f;
}

// ---------------------------------------------------------------------------
// Shared device bodies for the LSE GEMM.
// ---------------------------------------------------------------------------
__device__ __forceinline__ void lse_merge_lanes(float mRun[8], float sRun[8]) {
    #pragma unroll
    for (int r = 0; r < 8; ++r) {
        float m = mRun[r], s = sRun[r];
        #pragma unroll
        for (int off = 1; off < 16; off <<= 1) {
            float mo = __shfl_xor(m, off, 32);
            float so = __shfl_xor(s, off, 32);
            float mN = fmaxf(m, mo);
            s = s * __expf(m - mN) + so * __expf(mo - mN);
            m = mN;
        }
        mRun[r] = m; sRun[r] = s;
    }
}

__device__ __forceinline__ void lse_epilogue(float mRun[8], float sRun[8],
                                             float* red, int tid, int lane, int wid,
                                             int rowbase, float* __restrict__ lse_out) {
    lse_merge_lanes(mRun, sRun);
    if ((lane & 15) == 0) {
        int half = lane >> 4;
        #pragma unroll
        for (int r = 0; r < 8; ++r) {
            int lrow = half * 8 + r;
            red[(wid * 16 + lrow) * 2 + 0] = mRun[r];
            red[(wid * 16 + lrow) * 2 + 1] = sRun[r];
        }
    }
    __syncthreads();
    if (tid < TILE_M) {
        int row  = tid;
        int mW   = row >> 4;
        int lrow = row & 15;
        float m = -INFINITY, s = 0.0f;
        #pragma unroll
        for (int nw = 0; nw < 4; ++nw) {
            int w = nw * 2 + mW;
            float mo = red[(w * 16 + lrow) * 2 + 0];
            float so = red[(w * 16 + lrow) * 2 + 1];
            float mN = fmaxf(m, mo);
            s = s * __expf(m - mN) + so * __expf(mo - mN);
            m = mN;
        }
        lse_out[rowbase + row] = m + __logf(s);
    }
}

__device__ __forceinline__ void stage_x_tile(const float* __restrict__ X,
                                             unsigned short* As, int tid, int rowbase) {
    const float4* xb = (const float4*)(X + (size_t)rowbase * H_DIM);
    #pragma unroll 4
    for (int it = 0; it < 64; ++it) {
        int g   = it * 256 + tid;
        int row = g >> 9;
        int c4  = g & 511;
        float4 f = xb[(size_t)row * 512 + c4];
        unsigned int p0 = (unsigned int)f2bf_bits(f.x) | ((unsigned int)f2bf_bits(f.y) << 16);
        unsigned int p1 = (unsigned int)f2bf_bits(f.z) | ((unsigned int)f2bf_bits(f.w) << 16);
        unsigned int* dst = (unsigned int*)&As[(size_t)row * LDA_H + c4 * 4];
        dst[0] = p0;
        dst[1] = p1;
    }
}

__device__ __forceinline__ void load_frag_pair(BF16Frag& af, BF16Frag& bfv,
                                               const unsigned short* aBase,
                                               const unsigned short* wrow, int kt) {
    af.q[0]  = *(const uint4*)(aBase + kt);
    af.q[1]  = *(const uint4*)(aBase + kt + 16);
    bfv.q[0] = *(const uint4*)(wrow + kt);
    bfv.q[1] = *(const uint4*)(wrow + kt + 16);
}

// ---------------------------------------------------------------------------
// Kernel 2 (fast): fused logit GEMM + online logsumexp, W already bf16.
// Software-pipelined K-loop: ping-pong fragment registers so each WMMA
// overlaps the next stage's global/LDS loads (partial s_wait instead of 0).
// ---------------------------------------------------------------------------
__global__ __launch_bounds__(256)
void lse_gemm_bf16_kernel(const unsigned short* __restrict__ Wbf,
                          const float* __restrict__ X,
                          const float* __restrict__ bias,
                          float* __restrict__ lse_out) {
    extern __shared__ char smem[];
    unsigned short* As = (unsigned short*)smem;
    float* red = (float*)(smem + (size_t)TILE_M * LDA_H * sizeof(unsigned short));

    const int tid   = threadIdx.x;
    const int lane  = tid & 31;
    const int wid   = tid >> 5;
    const int mWave = wid & 1;
    const int nWave = wid >> 1;
    const int rowbase = blockIdx.x * TILE_M;

    stage_x_tile(X, As, tid, rowbase);
    __syncthreads();

    float mRun[8], sRun[8];
    #pragma unroll
    for (int r = 0; r < 8; ++r) { mRun[r] = -INFINITY; sRun[r] = 0.0f; }

    const int arow  = mWave * 16 + (lane & 15);
    const int kHalf = (lane & 16) ? 8 : 0;
    const unsigned short* aBase = &As[(size_t)arow * LDA_H + kHalf];

    for (int n0 = 0; n0 < V_DIM; n0 += TILE_N) {
        const int ncol = n0 + nWave * 16 + (lane & 15);
        const float bv = bias[ncol];
        const unsigned short* wrow = Wbf + (size_t)ncol * H_DIM + kHalf;

        f32x8 acc = {};
        BF16Frag a0, b0, a1, b1;

        // Prologue: load stage 0
        load_frag_pair(a0, b0, aBase, wrow, 0);

        #pragma unroll 1
        for (int kt = 0; kt < H_DIM - 64; kt += 64) {
            load_frag_pair(a1, b1, aBase, wrow, kt + 32);
            __builtin_prefetch(wrow + kt + 96, 0, 3);      // 2 K-steps ahead
            acc = __builtin_amdgcn_wmma_f32_16x16x32_bf16(
                      false, a0.v, false, b0.v, (short)0, acc, false, false);

            load_frag_pair(a0, b0, aBase, wrow, kt + 64);
            __builtin_prefetch(wrow + kt + 128, 0, 3);
            acc = __builtin_amdgcn_wmma_f32_16x16x32_bf16(
                      false, a1.v, false, b1.v, (short)0, acc, false, false);
        }
        // Epilogue: stages H-64 (in a0/b0) and H-32
        load_frag_pair(a1, b1, aBase, wrow, H_DIM - 32);
        acc = __builtin_amdgcn_wmma_f32_16x16x32_bf16(
                  false, a0.v, false, b0.v, (short)0, acc, false, false);
        acc = __builtin_amdgcn_wmma_f32_16x16x32_bf16(
                  false, a1.v, false, b1.v, (short)0, acc, false, false);

        #pragma unroll
        for (int r = 0; r < 8; ++r) {
            float x = acc[r] + bv;
            float mNew = fmaxf(mRun[r], x);
            sRun[r] = sRun[r] * __expf(mRun[r] - mNew) + __expf(x - mNew);
            mRun[r] = mNew;
        }
    }

    lse_epilogue(mRun, sRun, red, tid, lane, wid, rowbase, lse_out);
}

// ---------------------------------------------------------------------------
// Kernel 2 (fallback): same GEMM with on-the-fly W f32->bf16 (used only when
// the workspace is too small for the converted weight copy).
// ---------------------------------------------------------------------------
__global__ __launch_bounds__(256)
void lse_gemm_fallback_kernel(const float* __restrict__ W,
                              const float* __restrict__ X,
                              const float* __restrict__ bias,
                              float* __restrict__ lse_out) {
    extern __shared__ char smem[];
    unsigned short* As = (unsigned short*)smem;
    float* red = (float*)(smem + (size_t)TILE_M * LDA_H * sizeof(unsigned short));

    const int tid   = threadIdx.x;
    const int lane  = tid & 31;
    const int wid   = tid >> 5;
    const int mWave = wid & 1;
    const int nWave = wid >> 1;
    const int rowbase = blockIdx.x * TILE_M;

    stage_x_tile(X, As, tid, rowbase);
    __syncthreads();

    float mRun[8], sRun[8];
    #pragma unroll
    for (int r = 0; r < 8; ++r) { mRun[r] = -INFINITY; sRun[r] = 0.0f; }

    const int arow  = mWave * 16 + (lane & 15);
    const int kHalf = (lane & 16) ? 8 : 0;
    const unsigned short* aBase = &As[(size_t)arow * LDA_H + kHalf];

    for (int n0 = 0; n0 < V_DIM; n0 += TILE_N) {
        const int ncol = n0 + nWave * 16 + (lane & 15);
        const float bv = bias[ncol];
        const float* wrow = W + (size_t)ncol * H_DIM + kHalf;

        f32x8 acc = {};
        for (int kt = 0; kt < H_DIM; kt += 32) {
            BF16Frag af;
            af.q[0] = *(const uint4*)(aBase + kt);
            af.q[1] = *(const uint4*)(aBase + kt + 16);

            const float* wp = wrow + kt;
            float4 w0 = *(const float4*)(wp);
            float4 w1 = *(const float4*)(wp + 4);
            float4 w2 = *(const float4*)(wp + 16);
            float4 w3 = *(const float4*)(wp + 20);
            __builtin_prefetch(wp + 32, 0, 3);

            BF16Frag bfv;
            bfv.u[0]  = f2bf_bits(w0.x); bfv.u[1]  = f2bf_bits(w0.y);
            bfv.u[2]  = f2bf_bits(w0.z); bfv.u[3]  = f2bf_bits(w0.w);
            bfv.u[4]  = f2bf_bits(w1.x); bfv.u[5]  = f2bf_bits(w1.y);
            bfv.u[6]  = f2bf_bits(w1.z); bfv.u[7]  = f2bf_bits(w1.w);
            bfv.u[8]  = f2bf_bits(w2.x); bfv.u[9]  = f2bf_bits(w2.y);
            bfv.u[10] = f2bf_bits(w2.z); bfv.u[11] = f2bf_bits(w2.w);
            bfv.u[12] = f2bf_bits(w3.x); bfv.u[13] = f2bf_bits(w3.y);
            bfv.u[14] = f2bf_bits(w3.z); bfv.u[15] = f2bf_bits(w3.w);

            acc = __builtin_amdgcn_wmma_f32_16x16x32_bf16(
                      false, af.v, false, bfv.v, (short)0, acc, false, false);
        }

        #pragma unroll
        for (int r = 0; r < 8; ++r) {
            float x = acc[r] + bv;
            float mNew = fmaxf(mRun[r], x);
            sRun[r] = sRun[r] * __expf(mRun[r] - mNew) + __expf(x - mNew);
            mRun[r] = mNew;
        }
    }

    lse_epilogue(mRun, sRun, red, tid, lane, wid, rowbase, lse_out);
}

// ---------------------------------------------------------------------------
// Kernel 3: final CPO reduction (single block, deterministic).
// ---------------------------------------------------------------------------
__global__ __launch_bounds__(1024)
void cpo_reduce_kernel(const float* __restrict__ lse,
                       const float* __restrict__ tgt,
                       const int*   __restrict__ target,
                       float* __restrict__ out) {
    __shared__ float sred[1024];
    __shared__ float seqS[B2];
    __shared__ float seqC[B2];
    const int tid = threadIdx.x;

    for (int s = 0; s < B2; ++s) {
        int i = s * T_LEN + tid;
        int t = target[i];
        float mk = (t != IGNORE_INDEX) ? 1.0f : 0.0f;
        float pt = (tgt[i] - lse[i]) * mk;

        sred[tid] = pt;
        __syncthreads();
        for (int off = 512; off > 0; off >>= 1) {
            if (tid < off) sred[tid] += sred[tid + off];
            __syncthreads();
        }
        if (tid == 0) seqS[s] = sred[0];
        __syncthreads();

        sred[tid] = mk;
        __syncthreads();
        for (int off = 512; off > 0; off >>= 1) {
            if (tid < off) sred[tid] += sred[tid + off];
            __syncthreads();
        }
        if (tid == 0) seqC[s] = sred[0];
        __syncthreads();
    }

    if (tid == 0) {
        float num = 0.0f, den = 0.0f;
        for (int s = 0; s < B2 / 2; ++s) { num += seqS[s]; den += seqC[s]; }
        float nll = -num / den;

        float pref = 0.0f;
        for (int p = 0; p < B2 / 2; ++p) {
            float c = seqS[p] / seqC[p];
            float r = seqS[p + B2 / 2] / seqC[p + B2 / 2];
            float z = BETA_C * (c - r);
            pref += fminf(z, 0.0f) - log1pf(expf(-fabsf(z)));
        }
        pref *= (2.0f / (float)B2);

        out[0] = ALPHA_C * nll - pref;
    }
}

// ---------------------------------------------------------------------------
extern "C" void kernel_launch(void* const* d_in, const int* in_sizes, int n_in,
                              void* d_out, int out_size, void* d_ws, size_t ws_size,
                              hipStream_t stream) {
    const float* W      = (const float*)d_in[0];   // (V, H) f32
    const float* X      = (const float*)d_in[1];   // (B2, T, H) f32
    const int*   target = (const int*)  d_in[2];   // (B2, T) i32
    const float* bias   = (const float*)d_in[3];   // (V,) f32
    float* out = (float*)d_out;

    // Workspace layout: [lse: M floats][tgt: M floats][Wbf: V*H bf16]
    float* lse = (float*)d_ws;
    float* tgt = lse + M_TOK;
    const size_t small_bytes = (size_t)2 * M_TOK * sizeof(float);     // 64 KB
    unsigned short* Wbf = (unsigned short*)((char*)d_ws + small_bytes);
    const size_t need = small_bytes + (size_t)V_DIM * H_DIM * sizeof(unsigned short);

    // Kernel 1: target logits (1 wave per token)
    target_logit_kernel<<<M_TOK / 8, 256, 0, stream>>>(W, X, target, bias, tgt);

    const size_t smem_bytes = (size_t)TILE_M * LDA_H * sizeof(unsigned short)
                            + 8 * 16 * 2 * sizeof(float);

    if (ws_size >= need) {
        // Pre-convert W to bf16 (one pass, HBM-bound), then WMMA-bound GEMM.
        convert_w_kernel<<<8192, 256, 0, stream>>>(W, Wbf);
        (void)hipFuncSetAttribute((const void*)lse_gemm_bf16_kernel,
                                  hipFuncAttributeMaxDynamicSharedMemorySize,
                                  (int)smem_bytes);
        lse_gemm_bf16_kernel<<<M_TOK / TILE_M, 256, smem_bytes, stream>>>(Wbf, X, bias, lse);
    } else {
        (void)hipFuncSetAttribute((const void*)lse_gemm_fallback_kernel,
                                  hipFuncAttributeMaxDynamicSharedMemorySize,
                                  (int)smem_bytes);
        lse_gemm_fallback_kernel<<<M_TOK / TILE_M, 256, smem_bytes, stream>>>(W, X, bias, lse);
    }

    // Kernel 3: final scalar loss
    cpo_reduce_kernel<<<1, 1024, 0, stream>>>(lse, tgt, target, out);
}
